// Net_82626580840543
// MI455X (gfx1250) — matrix operand order repaired and checked by reading, not compiled, not run on hardware
//
#include <hip/hip_runtime.h>

typedef __attribute__((ext_vector_type(16))) _Float16 v16h;
typedef __attribute__((ext_vector_type(8)))  float    v8f;

#define BATCH    16384
#define SEQ      28
#define FEAT     28
#define DMODEL   64
#define DINNER   128
#define DSTATE   16
#define DTRANK   4
#define NLAYERS  6
#define NCLASSES 10

#define BTILE    16
#define ROWS     (BTILE * SEQ)        // 448 flattened (b,l) rows
#define MTILES   (ROWS / 16)          // 28
#define NTHREADS 256
#define NWAVES   (NTHREADS / 32)      // 8 wave32s

#define XDBW     36                   // dt_rank + 2*d_state

// LDS partition (element counts)
#define SHX_ELEMS   (ROWS * DINNER)   // 57344 halves
#define SHZ_ELEMS   (ROWS * DINNER)
#define SHH_ELEMS   (ROWS * DMODEL)   // 28672 halves
#define SHXDB_ELEMS (ROWS * XDBW)     // 16128 halves
#define SCRATCH_B   4096              // input_w f16 stage (64x32x2B) / pool (16x64 f32)
#define SMEM_BYTES  ((SHX_ELEMS + SHZ_ELEMS + SHH_ELEMS + SHXDB_ELEMS) * 2 + SCRATCH_B)
// = 318976 + 4096 = 323072 B  (<= 327680 B = 320 KB LDS/WGP)

// x tile per workgroup: 448 rows x 28 f32 contiguous = 50176 B = 3136 x 16 B
#define XTILE_CHUNKS (ROWS * FEAT / 4)

// ---- A fragment (16x32 f16, M x K) from LDS, contiguous K order (same as load_B_*):
// lanes 0-15: row M=lane, halves 0..15 = K 0..15
// lanes 16-31: row M=lane-16, halves 0..15 = K 16..31
__device__ __forceinline__ v16h load_A_lds(const _Float16* lds, int row0, int lda,
                                           int ko, int lane) {
  const int m  = lane & 15;
  const int kb = (lane & 16) ? 16 : 0;
  const _Float16* p = lds + (row0 + m) * lda + ko + kb;
  v16h a;
#pragma unroll
  for (int e = 0; e < 16; ++e) a[e] = p[e];
  return a;
}

// ---- B fragment (32x16 f16, K x N) from global f32 weights W[n][k] (row = out col):
// lanes 0-15 hold K 0..15, lanes 16-31 hold K 16..31; N = lane%16.
// Branchless: row index clamped (garbage columns are discarded by the store guard);
// 16 contiguous f32 per lane -> 4x global_load_b128.
__device__ __forceinline__ v16h load_B_gw(const float* W, int ldw, int n0, int ko,
                                          int lane, int nclamp) {
  int n = n0 + (lane & 15);
  n = (n > nclamp) ? nclamp : n;
  const int kb = ko + ((lane & 16) ? 16 : 0);
  const float* p = W + (size_t)n * ldw + kb;
  v16h b;
#pragma unroll
  for (int q = 0; q < 4; ++q) {
    const float4 f = *(const float4*)(p + 4 * q);
    b[4 * q + 0] = (_Float16)f.x;
    b[4 * q + 1] = (_Float16)f.y;
    b[4 * q + 2] = (_Float16)f.z;
    b[4 * q + 3] = (_Float16)f.w;
  }
  return b;
}

// ---- B fragment from pre-staged f16 LDS weights (ldw in halves), unguarded
__device__ __forceinline__ v16h load_B_lds(const _Float16* Wlds, int ldw, int n0,
                                           int ko, int lane) {
  const int n  = n0 + (lane & 15);
  const int kb = ko + ((lane & 16) ? 16 : 0);
  const _Float16* p = Wlds + n * ldw + kb;
  v16h b;
#pragma unroll
  for (int e = 0; e < 16; ++e) b[e] = p[e];
  return b;
}

// ---- D store: lane<16 -> N=lane, rows mb..mb+7 (mb=0); lane>=16 -> N=lane-16, mb=8
template<bool RES>
__device__ __forceinline__ void store_D(v8f c, _Float16* Dlds, int ldd, int row0,
                                        int n0, int nmax, const _Float16* Res,
                                        int ldr, int lane) {
  const int n  = n0 + (lane & 15);
  const int mb = (lane & 16) ? 8 : 0;
  if (n < nmax) {
#pragma unroll
    for (int j = 0; j < 8; ++j) {
      const int r = row0 + mb + j;
      float v = c[j];
      if (RES) v += (float)Res[r * ldr + n];
      Dlds[r * ldd + n] = (_Float16)v;
    }
  }
}

// ---- GEMM with f32 global weights: D(ROWS x nmax) = A(LDS f16) @ W^T (+Res)
template<int K, bool RES>
__device__ __forceinline__ void wmma_gemm_gw(
    const _Float16* __restrict__ Alds, int lda,
    const float* __restrict__ W, int ldw,
    int nTiles, int nmax,
    _Float16* __restrict__ Dlds, int ldd,
    const _Float16* __restrict__ Res, int ldr,
    int lane, int wave)
{
  for (int mt = wave; mt < MTILES; mt += NWAVES) {
    const int row0 = mt * 16;
    for (int nt = 0; nt < nTiles; ++nt) {
      const int n0 = nt * 16;
      v8f c = {};
#pragma unroll
      for (int ko = 0; ko < K; ko += 32) {
        v16h a = load_A_lds(Alds, row0, lda, ko, lane);
        v16h b = load_B_gw(W, ldw, n0, ko, lane, nmax - 1);
        c = __builtin_amdgcn_wmma_f32_16x16x32_f16(
                false, a, false, b, (short)0, c, false, false);
      }
      store_D<RES>(c, Dlds, ldd, row0, n0, nmax, Res, ldr, lane);
    }
  }
}

// ---- GEMM with f16 LDS-staged weights (input projection, K=32 incl. zero pad)
__device__ __forceinline__ void wmma_gemm_lw(
    const _Float16* __restrict__ Alds, int lda,
    const _Float16* __restrict__ Wlds, int ldw,
    int nTiles, int nmax,
    _Float16* __restrict__ Dlds, int ldd,
    int lane, int wave)
{
  for (int mt = wave; mt < MTILES; mt += NWAVES) {
    const int row0 = mt * 16;
    for (int nt = 0; nt < nTiles; ++nt) {
      const int n0 = nt * 16;
      v8f c = {};
      v16h a = load_A_lds(Alds, row0, lda, 0, lane);
      v16h b = load_B_lds(Wlds, ldw, n0, 0, lane);
      c = __builtin_amdgcn_wmma_f32_16x16x32_f16(
              false, a, false, b, (short)0, c, false, false);
      store_D<false>(c, Dlds, ldd, row0, n0, nmax, nullptr, 0, lane);
    }
  }
}

__device__ __forceinline__ float silu_f(float v) {
  return v / (1.0f + __expf(-v));
}

// sweep-prefetch a weight blob into cache (global_prefetch_b8)
__device__ __forceinline__ void prefetch_blob(const void* p, int bytes, int tid) {
  const char* c = (const char*)p;
  for (int off = tid * 128; off < bytes; off += NTHREADS * 128)
    __builtin_prefetch(c + off, 0, 1);
}

__global__ __launch_bounds__(NTHREADS)
void mamba_fused_kernel(
    const float* __restrict__ x,
    const float* __restrict__ input_w,
    const float* __restrict__ in_proj_w,
    const float* __restrict__ conv_w,
    const float* __restrict__ conv_b,
    const float* __restrict__ xproj_w,
    const float* __restrict__ dt_w,
    const float* __restrict__ dt_b,
    const float* __restrict__ A_log,
    const float* __restrict__ Dvec,
    const float* __restrict__ out_w,
    const float* __restrict__ cls_w,
    float* __restrict__ out)
{
  extern __shared__ char smem_raw[];
  _Float16* sh_x    = (_Float16*)smem_raw;            // xs / y   (448 x 128)
  _Float16* sh_z    = sh_x + SHX_ELEMS;               // z        (448 x 128); also x-stage
  _Float16* sh_h    = sh_z + SHZ_ELEMS;               // residual (448 x 64)
  _Float16* sh_xdb  = sh_h + SHH_ELEMS;               // dt|B|C   (448 x 36)
  char*     sh_scr  = (char*)(sh_xdb + SHXDB_ELEMS);  // 4 KB scratch
  _Float16* sh_iw   = (_Float16*)sh_scr;              // input_w staged (64 x 32 f16)
  float*    sh_pool = (float*)sh_scr;                 // pooled (16 x 64 f32), later

  const int tid    = threadIdx.x;
  const int lane   = tid & 31;
  const int wave   = tid >> 5;
  const int batch0 = blockIdx.x * BTILE;

  // ---- async-copy the contiguous x tile (50176 B) global -> LDS (sh_z region)
  // GLOBAL_LOAD_ASYNC_TO_LDS_B128, GVS mode: lds_off(v), goff32(v), base(s64)
  {
    const float* xbase = x + (size_t)batch0 * SEQ * FEAT;   // 16B-aligned (50176B/blk)
    const unsigned lds_base = (unsigned)(size_t)sh_z;       // 16B-aligned LDS offset
    for (int idx = tid; idx < XTILE_CHUNKS; idx += NTHREADS) {
      const unsigned off = (unsigned)idx * 16u;
      const unsigned lds_off = lds_base + off;
      asm volatile("global_load_async_to_lds_b128 %0, %1, %2"
                   :: "v"(lds_off), "v"(off), "s"(xbase)
                   : "memory");
    }
    asm volatile("s_wait_asynccnt 0x0" ::: "memory");
  }
  // ---- stage input_w (64x28 f32 -> 64x32 f16, zero-padded) into scratch
  for (int idx = tid; idx < DMODEL * 32; idx += NTHREADS) {
    const int n = idx >> 5, k = idx & 31;
    float v = 0.0f;
    if (k < FEAT) v = input_w[n * FEAT + k];
    sh_iw[idx] = (_Float16)v;
  }
  __syncthreads();

  // ---- convert staged f32 x tile -> f16 (ld=32, zero-pad K 28->32) in sh_x
  {
    const float* stage = (const float*)sh_z;
    for (int idx = tid; idx < ROWS * 32; idx += NTHREADS) {
      const int r = idx >> 5, k = idx & 31;
      float v = 0.0f;
      if (k < FEAT) v = stage[r * FEAT + k];
      sh_x[idx] = (_Float16)v;
    }
  }
  __syncthreads();

  // ---- input projection: (448,32pad) @ input_w^T -> sh_h (448,64)
  wmma_gemm_lw(sh_x, 32, sh_iw, 32, DMODEL / 16, DMODEL, sh_h, DMODEL, lane, wave);
  __syncthreads();

  for (int layer = 0; layer < NLAYERS; ++layer) {
    const float* Wi  = in_proj_w + (size_t)layer * 2 * DINNER * DMODEL;
    const float* Wxp = xproj_w   + (size_t)layer * XDBW * DINNER;
    const float* Wo  = out_w     + (size_t)layer * DMODEL * DINNER;

    // warm L2/WGP$ for this layer's weights
    prefetch_blob(Wi,  2 * DINNER * DMODEL * 4, tid);
    prefetch_blob(Wxp, XDBW * DINNER * 4, tid);
    prefetch_blob(Wo,  DMODEL * DINNER * 4, tid);

    // ---- in_proj: xs half (W rows 0..127) and z half (rows 128..255)
    wmma_gemm_gw<DMODEL, false>(sh_h, DMODEL, Wi, DMODEL,
                                DINNER / 16, DINNER, sh_x, DINNER,
                                nullptr, 0, lane, wave);
    wmma_gemm_gw<DMODEL, false>(sh_h, DMODEL, Wi + (size_t)DINNER * DMODEL, DMODEL,
                                DINNER / 16, DINNER, sh_z, DINNER,
                                nullptr, 0, lane, wave);
    __syncthreads();

    // ---- causal depthwise conv1d (k=3) + bias + SiLU, in place in sh_x
    for (int p = tid; p < BTILE * DINNER; p += NTHREADS) {
      const int b = p >> 7;
      const int d = p & (DINNER - 1);
      const size_t wdi = (size_t)layer * DINNER + d;
      const float w0 = conv_w[wdi * 3 + 0];
      const float w1 = conv_w[wdi * 3 + 1];
      const float w2 = conv_w[wdi * 3 + 2];
      const float cb = conv_b[wdi];
      float xm2 = 0.0f, xm1 = 0.0f;
      for (int l = 0; l < SEQ; ++l) {
        const int row = b * SEQ + l;
        const float x0 = (float)sh_x[row * DINNER + d];
        const float v  = silu_f(w0 * xm2 + w1 * xm1 + w2 * x0 + cb);
        sh_x[row * DINNER + d] = (_Float16)v;
        xm2 = xm1; xm1 = x0;
      }
    }
    __syncthreads();

    // ---- x_proj: (448,128) @ xproj_w^T(36,128) -> sh_xdb (store-guard N=36)
    wmma_gemm_gw<DINNER, false>(sh_x, DINNER, Wxp, DINNER,
                                3, XDBW, sh_xdb, XDBW,
                                nullptr, 0, lane, wave);
    __syncthreads();

    // ---- selective scan + delta(softplus) + SiLU(z) gating; y overwrites sh_x
    for (int p = tid; p < BTILE * DINNER; p += NTHREADS) {
      const int b = p >> 7;
      const int d = p & (DINNER - 1);
      const size_t di = (size_t)layer * DINNER + d;
      float Arow[DSTATE];
#pragma unroll
      for (int n = 0; n < DSTATE; ++n)
        Arow[n] = -__expf(A_log[di * DSTATE + n]);
      float dtw[DTRANK];
#pragma unroll
      for (int r = 0; r < DTRANK; ++r)
        dtw[r] = dt_w[di * DTRANK + r];
      const float dtb = dt_b[di];
      const float Dd  = Dvec[di];
      float st[DSTATE];
#pragma unroll
      for (int n = 0; n < DSTATE; ++n) st[n] = 0.0f;

      for (int l = 0; l < SEQ; ++l) {
        const int row = b * SEQ + l;
        const float u = (float)sh_x[row * DINNER + d];
        float dl = dtb;
#pragma unroll
        for (int r = 0; r < DTRANK; ++r)
          dl += (float)sh_xdb[row * XDBW + r] * dtw[r];
        const float delta = (dl > 20.0f) ? dl : __logf(1.0f + __expf(dl));
        float y = 0.0f;
#pragma unroll
        for (int n = 0; n < DSTATE; ++n) {
          const float Bv = (float)sh_xdb[row * XDBW + DTRANK + n];
          const float Cv = (float)sh_xdb[row * XDBW + DTRANK + DSTATE + n];
          const float dA = __expf(delta * Arow[n]);
          st[n] = dA * st[n] + delta * Bv * u;
          y += st[n] * Cv;
        }
        y += u * Dd;
        y *= silu_f((float)sh_z[row * DINNER + d]);
        sh_x[row * DINNER + d] = (_Float16)y;
      }
    }
    __syncthreads();

    // ---- out_proj + residual: sh_h = sh_h + y @ out_w^T
    wmma_gemm_gw<DINNER, true>(sh_x, DINNER, Wo, DINNER,
                               DMODEL / 16, DMODEL, sh_h, DMODEL,
                               sh_h, DMODEL, lane, wave);
    __syncthreads();
  }

  // ---- mean pool over seq (pool scratch reuses input_w staging area)
  for (int idx = tid; idx < BTILE * DMODEL; idx += NTHREADS) {
    const int b = idx >> 6, c = idx & (DMODEL - 1);
    float s = 0.0f;
    for (int l = 0; l < SEQ; ++l)
      s += (float)sh_h[(b * SEQ + l) * DMODEL + c];
    sh_pool[idx] = s * (1.0f / (float)SEQ);
  }
  __syncthreads();

  // ---- classifier (16 x 10)
  for (int idx = tid; idx < BTILE * NCLASSES; idx += NTHREADS) {
    const int b = idx / NCLASSES, cl = idx - b * NCLASSES;
    float s = 0.0f;
#pragma unroll
    for (int c = 0; c < DMODEL; ++c)
      s += sh_pool[b * DMODEL + c] * cls_w[cl * DMODEL + c];
    out[((size_t)(batch0 + b)) * NCLASSES + cl] = s;
  }
}

extern "C" void kernel_launch(void* const* d_in, const int* in_sizes, int n_in,
                              void* d_out, int out_size, void* d_ws, size_t ws_size,
                              hipStream_t stream) {
  (void)in_sizes; (void)n_in; (void)d_ws; (void)ws_size; (void)out_size;
  const float* x         = (const float*)d_in[0];
  const float* input_w   = (const float*)d_in[1];
  const float* in_proj_w = (const float*)d_in[2];
  const float* conv_w    = (const float*)d_in[3];
  const float* conv_b    = (const float*)d_in[4];
  const float* xproj_w   = (const float*)d_in[5];
  const float* dt_w      = (const float*)d_in[6];
  const float* dt_b      = (const float*)d_in[7];
  const float* A_log     = (const float*)d_in[8];
  const float* Dvec      = (const float*)d_in[9];
  const float* out_w     = (const float*)d_in[10];
  const float* cls_w     = (const float*)d_in[11];
  float* out = (float*)d_out;

  dim3 grid(BATCH / BTILE);   // 1024 workgroups, one 16-batch tile each
  dim3 block(NTHREADS);       // 8 wave32s
  hipLaunchKernelGGL(mamba_fused_kernel, grid, block, SMEM_BYTES, stream,
                     x, input_w, in_proj_w, conv_w, conv_b, xproj_w, dt_w, dt_b,
                     A_log, Dvec, out_w, cls_w, out);
}
